// GRUCell_81157702025454
// MI455X (gfx1250) — matrix operand order repaired and checked by reading, not compiled, not run on hardware
//
#include <hip/hip_runtime.h>
#include <hip/hip_bf16.h>

// ---------------------------------------------------------------------------
// Fused GRU cell for MI455X (gfx1250, wave32, WMMA bf16 -> f32 accumulate)
//   B=32768, IN=256, H=512, BB=4
// Packed weight layout: Wp[gate*512 + col][k], k in [0,800):
//   [0,256)  = W_ih row   (K over x)
//   [256,768)= W_hh row   (K over hx)
//   [768,772)= W_vel row  (K over vel)   [772,800) = zero pad
// K loop split into three statically-typed phases so the n-gate accumulator
// target (x_n+v_n vs h_n) is fixed per phase. Double-buffered LDS, one
// barrier per K-step.
// ---------------------------------------------------------------------------

typedef __attribute__((ext_vector_type(16))) __bf16 bf16x16;
typedef __attribute__((ext_vector_type(8)))  float  floatx8;

#define WMMA_BF16(A, B, C) \
  __builtin_amdgcn_wmma_f32_16x16x32_bf16(false, (A), false, (B), (short)0, (C), false, false)

__device__ __forceinline__ unsigned short f32_to_bf16(float f) {
  unsigned int u = __float_as_uint(f);
  unsigned int r = u + 0x7fffu + ((u >> 16) & 1u);   // round-to-nearest-even
  return (unsigned short)(r >> 16);
}

// Pack two f32 into one dword of 2x bf16:
//   round (add 0x8000: nearest, ties-away) then v_perm_b32 grabs the two
//   high half-words in a single instruction.  3 VALU ops per 2 elements.
__device__ __forceinline__ unsigned int pack_bf16x2(float lo, float hi) {
  unsigned int ul = __float_as_uint(lo) + 0x8000u;
  unsigned int uh = __float_as_uint(hi) + 0x8000u;
  // dst = { uh[31:24], uh[23:16], ul[31:24], ul[23:16] }
  return __builtin_amdgcn_perm(uh, ul, 0x07060302u);
}

struct FragU {
  union { uint4 q[2]; bf16x16 v; };
};

// A-fragment (16x32 bf16): lanes 0-15 row M=lane hold K{0..7,16..23},
// lanes 16-31 row M=lane-16 hold K{8..15,24..31}  (ISA 7.12.2 table)
__device__ __forceinline__ void ldsA(FragU& f, const unsigned short* base,
                                     int rowBase, int lane) {
  const unsigned short* p = base + (rowBase + (lane & 15)) * 40;
  int h8 = (lane >> 4) * 8;
  f.q[0] = *(const uint4*)(p + h8);
  f.q[1] = *(const uint4*)(p + 16 + h8);
}

// B-fragment (32x16 bf16, stored as [n][k] in LDS): lanes 0-15 col N=lane
// hold K{0..7,16..23}; lanes 16-31 hold K{8..15,24..31} (matches A)
__device__ __forceinline__ void ldsB(FragU& f, const unsigned short* base,
                                     int rowBase, int lane) {
  const unsigned short* p = base + (rowBase + (lane & 15)) * 40;
  int h16 = (lane >> 4) * 8;
  f.q[0] = *(const uint4*)(p + h16);
  f.q[1] = *(const uint4*)(p + h16 + 16);
}

// ----------------------------- prepass kernels -----------------------------

__global__ __launch_bounds__(256) void pack_weights(const float* __restrict__ W_ih,
                                                    const float* __restrict__ W_hh,
                                                    const float* __restrict__ W_vel,
                                                    unsigned short* __restrict__ Wp) {
  int n = blockIdx.x;           // 0..1535
  int t = threadIdx.x;          // 0..255
  unsigned short* dst = Wp + (size_t)n * 800;
  dst[t]        = f32_to_bf16(W_ih[(size_t)n * 256 + t]);
  dst[256 + t]  = f32_to_bf16(W_hh[(size_t)n * 512 + t]);
  dst[512 + t]  = f32_to_bf16(W_hh[(size_t)n * 512 + 256 + t]);
  if (t < 32) dst[768 + t] = (t < 4) ? f32_to_bf16(W_vel[n * 4 + t]) : (unsigned short)0;
}

__global__ __launch_bounds__(256) void pack_bias(const float* __restrict__ b_ih,
                                                 const float* __restrict__ b_hh,
                                                 const float* __restrict__ b_vel,
                                                 float* __restrict__ biasR,
                                                 float* __restrict__ biasU,
                                                 float* __restrict__ biasNX,
                                                 float* __restrict__ biasNH) {
  int c = blockIdx.x * blockDim.x + threadIdx.x;
  if (c < 512) {
    biasR[c]  = b_ih[c] + b_hh[c] + b_vel[c];
    biasU[c]  = b_ih[512 + c] + b_hh[512 + c] + b_vel[512 + c];
    biasNX[c] = b_ih[1024 + c] + b_vel[1024 + c];
    biasNH[c] = b_hh[1024 + c];
  }
}

// ------------------------------- main kernel -------------------------------
// grid: (H/32, B/128) ; block: 256 threads = 8 waves
// wave (wm = w>>1, wn = w&1): rows wm*32..+32, cols wn*16..+16 (per gate)

__global__ __launch_bounds__(256) void gru_fused(const float* __restrict__ x,
                                                 const float* __restrict__ hx,
                                                 const float* __restrict__ vel,
                                                 const unsigned short* __restrict__ Wp,
                                                 const float* __restrict__ biasR,
                                                 const float* __restrict__ biasU,
                                                 const float* __restrict__ biasNX,
                                                 const float* __restrict__ biasNH,
                                                 float* __restrict__ hy) {
  __shared__ unsigned short As[2][128][40];     // 2 x (128 rows x 32 K, pad->40)
  __shared__ unsigned short Bs[2][3][32][40];   // 2 x (3 gates x 32 n x 32 K)

  const int tid   = threadIdx.x;
  const int lane  = tid & 31;
  const int wv    = tid >> 5;
  const int wm    = wv >> 1;                 // 0..3 : row group
  const int wn    = wv & 1;                  // 0..1 : col group
  const int bN    = blockIdx.x;              // H tile (32 cols)
  const int bM    = blockIdx.y;              // batch tile (128 rows)

  floatx8 acc_r[2]  = {};
  floatx8 acc_u[2]  = {};
  floatx8 acc_nx[2] = {};                    // x_n + v_n part
  floatx8 acc_nh[2] = {};                    // h_n part

  const int arow = tid >> 1;                 // 0..127
  const int akb  = (tid & 1) * 16;           // 0 or 16
  const size_t rGa = (size_t)bM * 128 + arow;

  // B staging role: threads 0..191, 32 bf16 halves each
  const int brr   = tid >> 1;
  const int bhalf = tid & 1;
  const int bgate = brr >> 5;
  const int bnn   = brr & 31;
  const unsigned short* bSrc =
      Wp + (size_t)(bgate * 512 + bN * 32 + bnn) * 800 + bhalf * 16;

  // ---- staging helpers ----
  auto cvt16 = [&](int buf, const float* __restrict__ src) {
    float4 f0 = *(const float4*)(src + 0);
    float4 f1 = *(const float4*)(src + 4);
    float4 f2 = *(const float4*)(src + 8);
    float4 f3 = *(const float4*)(src + 12);
    union { unsigned int d[8]; uint4 q[2]; } pk;
    pk.d[0] = pack_bf16x2(f0.x, f0.y);
    pk.d[1] = pack_bf16x2(f0.z, f0.w);
    pk.d[2] = pack_bf16x2(f1.x, f1.y);
    pk.d[3] = pack_bf16x2(f1.z, f1.w);
    pk.d[4] = pack_bf16x2(f2.x, f2.y);
    pk.d[5] = pack_bf16x2(f2.z, f2.w);
    pk.d[6] = pack_bf16x2(f3.x, f3.y);
    pk.d[7] = pack_bf16x2(f3.z, f3.w);
    *(uint4*)&As[buf][arow][akb]     = pk.q[0];
    *(uint4*)&As[buf][arow][akb + 8] = pk.q[1];
  };
  auto stageB = [&](int buf, int gk) {
    if (tid < 192) {
      const unsigned short* src = bSrc + gk;
      uint4 q0 = *(const uint4*)(src);
      uint4 q1 = *(const uint4*)(src + 8);
      *(uint4*)&Bs[buf][bgate][bnn][bhalf * 16]     = q0;
      *(uint4*)&Bs[buf][bgate][bnn][bhalf * 16 + 8] = q1;
    }
  };
  auto stage_x = [&](int buf, int kk) {          // kk in [0,256)
    cvt16(buf, x + rGa * 256 + kk + akb);
    stageB(buf, kk);
  };
  auto stage_h = [&](int buf, int kk) {          // kk in [0,512)
    cvt16(buf, hx + rGa * 512 + kk + akb);
    stageB(buf, 256 + kk);
  };
  auto stage_v = [&](int buf) {                  // K chunk 768..799 (vel + pad)
    float4 f0 = make_float4(0.f, 0.f, 0.f, 0.f);
    if (akb == 0) f0 = *(const float4*)(vel + rGa * 4);
    union { unsigned int d[8]; uint4 q[2]; } pk;
    pk.d[0] = pack_bf16x2(f0.x, f0.y);
    pk.d[1] = pack_bf16x2(f0.z, f0.w);
    #pragma unroll
    for (int j = 2; j < 8; ++j) pk.d[j] = 0u;
    *(uint4*)&As[buf][arow][akb]     = pk.q[0];
    *(uint4*)&As[buf][arow][akb + 8] = pk.q[1];
    stageB(buf, 768);
  };

  // one K-step: load frags from BUF, run STAGE for next buffer, 6 WMMAs
#define GRU_STEP(BUF, ACCN, STAGE_STMT)                                   \
  {                                                                       \
    FragU a0, a1, fbr, fbu, fbn;                                          \
    ldsA(a0, &As[BUF][0][0], wm * 32,      lane);                         \
    ldsA(a1, &As[BUF][0][0], wm * 32 + 16, lane);                         \
    ldsB(fbr, &Bs[BUF][0][0][0], wn * 16, lane);                          \
    ldsB(fbu, &Bs[BUF][1][0][0], wn * 16, lane);                          \
    ldsB(fbn, &Bs[BUF][2][0][0], wn * 16, lane);                          \
    STAGE_STMT;                                                           \
    acc_r[0] = WMMA_BF16(a0.v, fbr.v, acc_r[0]);                          \
    acc_r[1] = WMMA_BF16(a1.v, fbr.v, acc_r[1]);                          \
    acc_u[0] = WMMA_BF16(a0.v, fbu.v, acc_u[0]);                          \
    acc_u[1] = WMMA_BF16(a1.v, fbu.v, acc_u[1]);                          \
    ACCN[0]  = WMMA_BF16(a0.v, fbn.v, ACCN[0]);                           \
    ACCN[1]  = WMMA_BF16(a1.v, fbn.v, ACCN[1]);                           \
    __syncthreads();                                                      \
  }

  // prologue
  stage_x(0, 0);
  __syncthreads();
  int cur = 0;

  // phase 1: x region (K 0..255), n-gate -> acc_nx
  #pragma unroll 1
  for (int c = 0; c < 8; ++c) {
    int nxt = cur ^ 1;
    GRU_STEP(cur, acc_nx, {
      if (c < 7) stage_x(nxt, (c + 1) * 32);
      else       stage_h(nxt, 0);
    });
    cur = nxt;
  }
  // phase 2: hx region (K 256..767), n-gate -> acc_nh
  #pragma unroll 1
  for (int c = 0; c < 16; ++c) {
    int nxt = cur ^ 1;
    GRU_STEP(cur, acc_nh, {
      if (c < 15) stage_h(nxt, (c + 1) * 32);
      else        stage_v(nxt);
    });
    cur = nxt;
  }
  // phase 3: vel chunk (K 768..799), n-gate -> acc_nx
  GRU_STEP(cur, acc_nx, {});

#undef GRU_STEP

  // ---- fused GRU epilogue ----
  // C/D layout: lanes 0-15 -> N=lane, M=i ; lanes 16-31 -> N=lane-16, M=8+i
  const int nl  = lane & 15;
  const int hh  = lane >> 4;
  const int col = bN * 32 + wn * 16 + nl;
  const float bR  = biasR[col];
  const float bU  = biasU[col];
  const float bNX = biasNX[col];
  const float bNH = biasNH[col];

  #pragma unroll
  for (int s = 0; s < 2; ++s) {
    const int mbase = bM * 128 + wm * 32 + s * 16 + hh * 8;
    #pragma unroll
    for (int i = 0; i < 8; ++i) {
      const size_t row = (size_t)(mbase + i);
      float r  = acc_r[s][i]  + bR;
      float u  = acc_u[s][i]  + bU;
      float xn = acc_nx[s][i] + bNX;
      float hn = acc_nh[s][i] + bNH;
      float rg = 1.0f / (1.0f + __expf(-r));
      float ug = 1.0f / (1.0f + __expf(-u));
      float z  = __expf(-2.0f * (xn + rg * hn));
      float ng = (1.0f - z) / (1.0f + z);      // tanh
      float h0 = hx[row * 512 + col];
      hy[row * 512 + col] = ug * h0 + (1.0f - ug) * ng;
    }
  }
}

// ------------------------------- launcher ----------------------------------

extern "C" void kernel_launch(void* const* d_in, const int* in_sizes, int n_in,
                              void* d_out, int out_size, void* d_ws, size_t ws_size,
                              hipStream_t stream) {
  const float* x     = (const float*)d_in[0];
  const float* hx    = (const float*)d_in[1];
  const float* vel   = (const float*)d_in[2];
  const float* W_ih  = (const float*)d_in[3];
  const float* b_ih  = (const float*)d_in[4];
  const float* W_hh  = (const float*)d_in[5];
  const float* b_hh  = (const float*)d_in[6];
  const float* W_vel = (const float*)d_in[7];
  const float* b_vel = (const float*)d_in[8];
  float* hy = (float*)d_out;

  // workspace layout
  unsigned short* Wp = (unsigned short*)d_ws;            // 1536*800*2 = 2,457,600 B
  char* base = (char*)d_ws + 1536 * 800 * sizeof(unsigned short);
  float* biasR  = (float*)(base + 0 * 512 * sizeof(float));
  float* biasU  = (float*)(base + 1 * 512 * sizeof(float));
  float* biasNX = (float*)(base + 2 * 512 * sizeof(float));
  float* biasNH = (float*)(base + 3 * 512 * sizeof(float));

  pack_weights<<<1536, 256, 0, stream>>>(W_ih, W_hh, W_vel, Wp);
  pack_bias<<<2, 256, 0, stream>>>(b_ih, b_hh, b_vel, biasR, biasU, biasNX, biasNH);

  dim3 grid(512 / 32, 32768 / 128);   // (16, 256)
  gru_fused<<<grid, 256, 0, stream>>>(x, hx, vel, Wp, biasR, biasU, biasNX, biasNH, hy);
}
